// GNNSetRepClassifierSubstruct_62938450755892
// MI455X (gfx1250) — matrix operand order repaired and checked by reading, not compiled, not run on hardware
//
#include <hip/hip_runtime.h>

typedef _Float16 h16_t;
typedef __attribute__((ext_vector_type(16))) _Float16 v16h;
typedef __attribute__((ext_vector_type(8)))  _Float16 v8h;
typedef __attribute__((ext_vector_type(8)))  float    v8f;

#define N_NODES 204800
#define N_EDGES 409600
#define IN_CH   128
#define HID     256
#define NCOMP   8192
#define NPC     25
#define BATCH   4096
#define SMDIM   512   // S*M
#define SDIM    64
#define MDIM    8
#define BN_EPS  1e-5f

#define SHUF16(lo, hi) __builtin_shufflevector(lo, hi, 0,1,2,3,4,5,6,7,8,9,10,11,12,13,14,15)
#define WMMA(a, b, c) __builtin_amdgcn_wmma_f32_16x16x32_f16(false, a, false, b, (short)0, c, false, false)

// ---------------------------------------------------------------------------
// Weight packing: f32 [K,Ko] row-major -> f16 fragment-order so the GEMM's
// B-operand loads are contiguous b128s.
// Packed index = ((kt*(Ko/16) + ct)*32 + lane)*16 + j, where element j of the
// v16h B fragment holds K = kt*32 + 16*(lane>>4) + j, col = ct*16 + (lane&15).
// ---------------------------------------------------------------------------
__global__ void pack_w_kernel(const float* __restrict__ W, h16_t* __restrict__ Wp,
                              int K, int Ko) {
    int idx = blockIdx.x * blockDim.x + threadIdx.x;
    int total = K * Ko;
    if (idx >= total) return;
    int j    = idx & 15;
    int lane = (idx >> 4) & 31;
    int tile = idx >> 9;
    int ct_count = Ko >> 4;
    int kt = tile / ct_count;
    int ct = tile - kt * ct_count;
    int hf = lane >> 4, r = lane & 15;
    int kk = hf * 16 + j;                       // B-matrix K striping
    Wp[idx] = (h16_t)W[(size_t)(kt * 32 + kk) * Ko + ct * 16 + r];
}

__global__ void zerof_kernel(float* __restrict__ p, int n) {
    int i = blockIdx.x * blockDim.x + threadIdx.x;
    if (i < n) p[i] = 0.0f;
}

// ---------------------------------------------------------------------------
// Edge scatter-add. Layer 0: f32 x, no activation. Layers 1..: f16 h, relu.
// ---------------------------------------------------------------------------
__global__ void gather0_kernel(const float* __restrict__ x, const int* __restrict__ src,
                               const int* __restrict__ dst, float* __restrict__ agg) {
    int idx = blockIdx.x * blockDim.x + threadIdx.x;   // E * 32 threads, 4 ch each
    int e  = idx >> 5;
    int c4 = (idx & 31) * 4;
    if (e >= N_EDGES) return;
    int s = src[e], d = dst[e];
    float4 v = *(const float4*)(x + (size_t)s * IN_CH + c4);
    float* ap = agg + (size_t)d * IN_CH + c4;
    atomicAdd(ap + 0, v.x); atomicAdd(ap + 1, v.y);
    atomicAdd(ap + 2, v.z); atomicAdd(ap + 3, v.w);
}

__global__ void gather_h_kernel(const h16_t* __restrict__ h, const int* __restrict__ src,
                                const int* __restrict__ dst, float* __restrict__ agg) {
    int idx = blockIdx.x * blockDim.x + threadIdx.x;   // E * 32 threads, 8 ch each
    int e  = idx >> 5;
    int c8 = (idx & 31) * 8;
    if (e >= N_EDGES) return;
    int s = src[e], d = dst[e];
    v8h v = *(const v8h*)(h + (size_t)s * HID + c8);
    float* ap = agg + (size_t)d * HID + c8;
#pragma unroll
    for (int i = 0; i < 8; ++i) {
        float f = (float)v[i];
        f = f > 0.0f ? f : 0.0f;                       // relu(h) message
        atomicAdd(ap + i, f);
    }
}

// m = x + agg  -> f16
__global__ void combine0_kernel(const float* __restrict__ x, const float* __restrict__ agg,
                                h16_t* __restrict__ a16) {
    int i = blockIdx.x * blockDim.x + threadIdx.x;
    if (i < N_NODES * IN_CH) a16[i] = (h16_t)(x[i] + agg[i]);
}

// m = relu(h) + agg -> f16
__global__ void combine_h_kernel(const h16_t* __restrict__ h, const float* __restrict__ agg,
                                 h16_t* __restrict__ a16) {
    int i = blockIdx.x * blockDim.x + threadIdx.x;
    if (i >= N_NODES * HID) return;
    float f = (float)h[i];
    f = f > 0.0f ? f : 0.0f;
    a16[i] = (h16_t)(f + agg[i]);
}

// ---------------------------------------------------------------------------
// WMMA f16 GEMM: out[M,Ko] = act(A[M,K] @ W[K,Ko] + bias)
// One wave computes a 32-row x 64-col tile: 2 A fragments x 4 B fragments ->
// 8 WMMAs per k-chunk on 12 b128 loads (all loads issued before the WMMA
// chain so the scheduler can use partial loadcnt waits).
// act: 0=none, 1=relu, 2=leaky(0.01). out16 xor out32 non-null.
// ---------------------------------------------------------------------------
__global__ void wmma_gemm_kernel(const h16_t* __restrict__ A, const h16_t* __restrict__ Wp,
                                 const float* __restrict__ bias,
                                 h16_t* __restrict__ out16, float* __restrict__ out32,
                                 int M, int K, int Ko, int act) {
    int wid  = (blockIdx.x * blockDim.x + threadIdx.x) >> 5;
    int lane = threadIdx.x & 31;
    int ntc  = Ko >> 6;                      // 64-col tiles
    int strip = wid / ntc;
    if (strip >= (M >> 5)) return;           // 32-row strips
    int cb   = (wid - strip * ntc) << 6;
    int row0 = strip << 5;
    int hf = lane >> 4, r = lane & 15;

    const h16_t* arow0 = A + (size_t)(row0 + r) * K;
    const h16_t* arow1 = A + (size_t)(row0 + 16 + r) * K;
    int ct_count = Ko >> 4;
    int ct0 = cb >> 4;

    v8f acc[8] = {};
    for (int k0 = 0; k0 < K; k0 += 32) {
        // --- issue all fragment loads first (A then B) ---
        v8h a0lo = *(const v8h*)(arow0 + k0 + hf * 16);
        v8h a0hi = *(const v8h*)(arow0 + k0 + hf * 16 + 8);
        v8h a1lo = *(const v8h*)(arow1 + k0 + hf * 16);
        v8h a1hi = *(const v8h*)(arow1 + k0 + hf * 16 + 8);
        int kt = k0 >> 5;
        const h16_t* wb = Wp + (((size_t)kt * ct_count + ct0) * 32 + lane) * 16;
        v16h b[4];
#pragma unroll
        for (int t = 0; t < 4; ++t) {
            v8h blo = *(const v8h*)(wb + t * 512);
            v8h bhi = *(const v8h*)(wb + t * 512 + 8);
            b[t] = SHUF16(blo, bhi);
        }
        v16h a0 = SHUF16(a0lo, a0hi);
        v16h a1 = SHUF16(a1lo, a1hi);
        // --- WMMA chain: each B fragment reused for both row halves ---
#pragma unroll
        for (int t = 0; t < 4; ++t) {
            acc[t]     = WMMA(a0, b[t], acc[t]);
            acc[4 + t] = WMMA(a1, b[t], acc[4 + t]);
        }
    }

    // Epilogue: C/D layout -> row = rr + 8*hf (+16 for upper half), col = cb + t*16 + r
#pragma unroll
    for (int rh = 0; rh < 2; ++rh) {
#pragma unroll
        for (int t = 0; t < 4; ++t) {
            int col = cb + t * 16 + r;
            float bv = bias ? bias[col] : 0.0f;
#pragma unroll
            for (int rr = 0; rr < 8; ++rr) {
                int m = rh * 16 + hf * 8 + rr;
                float v = acc[rh * 4 + t][rr] + bv;
                if (act == 1)      v = v > 0.0f ? v : 0.0f;
                else if (act == 2) v = v > 0.0f ? v : 0.01f * v;
                size_t o = (size_t)(row0 + m) * Ko + col;
                if (out16) out16[o] = (h16_t)v;
                else       out32[o] = v;
            }
        }
    }
}

// ---------------------------------------------------------------------------
// Component mean pooling: 25 contiguous nodes per component, divisor = 25
// (all component sizes equal -> per-graph max component size == 25).
// ---------------------------------------------------------------------------
__global__ void pool_kernel(const h16_t* __restrict__ h, h16_t* __restrict__ Xp) {
    int idx = blockIdx.x * blockDim.x + threadIdx.x;   // NCOMP * HID
    if (idx >= NCOMP * HID) return;
    int c = idx >> 8;
    int j = idx & 255;
    const h16_t* p = h + (size_t)c * NPC * HID + j;
    float s = 0.0f;
#pragma unroll
    for (int n = 0; n < NPC; ++n) s += (float)p[n * HID];
    Xp[idx] = (h16_t)(s * (1.0f / NPC));
}

// max over MDIM chunks of SDIM, sum over 2 components
__global__ void setred_kernel(const float* __restrict__ t, float* __restrict__ ts) {
    int idx = blockIdx.x * blockDim.x + threadIdx.x;   // BATCH * SDIM
    if (idx >= BATCH * SDIM) return;
    int b = idx >> 6;
    int s = idx & 63;
    float acc = 0.0f;
#pragma unroll
    for (int c = 0; c < 2; ++c) {
        const float* row = t + (size_t)(b * 2 + c) * SMDIM;
        float mx = row[s];
#pragma unroll
        for (int m = 1; m < MDIM; ++m) mx = fmaxf(mx, row[m * SDIM + s]);
        acc += mx;
    }
    ts[idx] = acc;
}

__global__ void fc1_kernel(const float* __restrict__ ts, const float* __restrict__ w,
                           const float* __restrict__ b, float* __restrict__ t1) {
    int idx = blockIdx.x * blockDim.x + threadIdx.x;   // BATCH * 32
    if (idx >= BATCH * 32) return;
    int bb = idx >> 5;
    int o  = idx & 31;
    const float* row = ts + (size_t)bb * 64;
    float acc = b[o];
#pragma unroll
    for (int k = 0; k < 64; ++k) acc += row[k] * w[k * 32 + o];
    t1[idx] = acc;
}

__global__ void bnstats_kernel(const float* __restrict__ t1, float* __restrict__ stats) {
    int o = blockIdx.x;                                // 32 blocks, one per channel
    __shared__ float ssum[256], ssq[256];
    float s = 0.0f, q = 0.0f;
    for (int b = threadIdx.x; b < BATCH; b += 256) {
        float v = t1[b * 32 + o];
        s += v; q += v * v;
    }
    ssum[threadIdx.x] = s; ssq[threadIdx.x] = q;
    __syncthreads();
    for (int st = 128; st > 0; st >>= 1) {
        if (threadIdx.x < st) {
            ssum[threadIdx.x] += ssum[threadIdx.x + st];
            ssq[threadIdx.x]  += ssq[threadIdx.x + st];
        }
        __syncthreads();
    }
    if (threadIdx.x == 0) {
        float mu  = ssum[0] / (float)BATCH;
        float var = ssq[0] / (float)BATCH - mu * mu;   // population variance
        stats[o] = mu; stats[32 + o] = var;
    }
}

__global__ void head_kernel(const float* __restrict__ t1, const float* __restrict__ stats,
                            const float* __restrict__ g, const float* __restrict__ be,
                            const float* __restrict__ w2, const float* __restrict__ b2,
                            float* __restrict__ out) {
    int b = blockIdx.x * blockDim.x + threadIdx.x;
    if (b >= BATCH) return;
    float l0 = b2[0], l1 = b2[1];
#pragma unroll
    for (int o = 0; o < 32; ++o) {
        float v = (t1[b * 32 + o] - stats[o]) * rsqrtf(stats[32 + o] + BN_EPS) * g[o] + be[o];
        v = v > 0.0f ? v : 0.01f * v;                  // leaky relu
        l0 += v * w2[o * 2 + 0];
        l1 += v * w2[o * 2 + 1];
    }
    float mx  = fmaxf(l0, l1);
    float lse = mx + logf(expf(l0 - mx) + expf(l1 - mx));
    out[b * 2 + 0] = l0 - lse;
    out[b * 2 + 1] = l1 - lse;
}

// ---------------------------------------------------------------------------
extern "C" void kernel_launch(void* const* d_in, const int* in_sizes, int n_in,
                              void* d_out, int out_size, void* d_ws, size_t ws_size,
                              hipStream_t stream) {
    const float* x     = (const float*)d_in[0];
    const float* w0a   = (const float*)d_in[1];
    const float* b0a   = (const float*)d_in[2];
    const float* w0b   = (const float*)d_in[3];
    const float* b0b   = (const float*)d_in[4];
    const float* wra   = (const float*)d_in[5];
    const float* bra   = (const float*)d_in[6];
    const float* wrb   = (const float*)d_in[7];
    const float* brb   = (const float*)d_in[8];
    const float* Wc    = (const float*)d_in[9];
    const float* fc1_w = (const float*)d_in[10];
    const float* fc1_b = (const float*)d_in[11];
    const float* bn_g  = (const float*)d_in[12];
    const float* bn_b  = (const float*)d_in[13];
    const float* fc2_w = (const float*)d_in[14];
    const float* fc2_b = (const float*)d_in[15];
    const int*   ei    = (const int*)d_in[16];
    const int*   src   = ei;
    const int*   dst   = ei + N_EDGES;
    float* out = (float*)d_out;

    // workspace carve-up
    char* ws = (char*)d_ws;
    size_t off = 0;
    auto carve = [&](size_t bytes) -> char* {
        char* p = ws + off;
        off += (bytes + 255) & ~((size_t)255);
        return p;
    };
    h16_t* h16   = (h16_t*)carve((size_t)N_NODES * HID * 2);
    h16_t* a16   = (h16_t*)carve((size_t)N_NODES * HID * 2);
    h16_t* b16   = (h16_t*)carve((size_t)N_NODES * HID * 2);
    float* agg   = (float*)carve((size_t)N_NODES * HID * 4);
    h16_t* wpool = (h16_t*)carve((size_t)491520 * 2);
    h16_t* Xp    = (h16_t*)carve((size_t)NCOMP * HID * 2);
    float* tcls  = (float*)carve((size_t)NCOMP * SMDIM * 4);
    float* tsum  = (float*)carve((size_t)BATCH * SDIM * 4);
    float* t1    = (float*)carve((size_t)BATCH * 32 * 4);
    float* stats = (float*)carve(64 * 4);

    h16_t* w0a_p  = wpool;            // 128*256
    h16_t* w0b_p  = wpool + 32768;    // 256*256
    h16_t* wra0_p = wpool + 98304;
    h16_t* wra1_p = wpool + 163840;
    h16_t* wrb0_p = wpool + 229376;
    h16_t* wrb1_p = wpool + 294912;
    h16_t* Wc_p   = wpool + 360448;   // 256*512

    const int TB = 256;
    auto cdiv = [](int a, int b) { return (a + b - 1) / b; };

    // ---- weight packing (one pass, tiny) ----
    pack_w_kernel<<<cdiv(IN_CH * HID, TB), TB, 0, stream>>>(w0a, w0a_p, IN_CH, HID);
    pack_w_kernel<<<cdiv(HID * HID, TB), TB, 0, stream>>>(w0b, w0b_p, HID, HID);
    pack_w_kernel<<<cdiv(HID * HID, TB), TB, 0, stream>>>(wra,              wra0_p, HID, HID);
    pack_w_kernel<<<cdiv(HID * HID, TB), TB, 0, stream>>>(wra + HID * HID,  wra1_p, HID, HID);
    pack_w_kernel<<<cdiv(HID * HID, TB), TB, 0, stream>>>(wrb,              wrb0_p, HID, HID);
    pack_w_kernel<<<cdiv(HID * HID, TB), TB, 0, stream>>>(wrb + HID * HID,  wrb1_p, HID, HID);
    pack_w_kernel<<<cdiv(HID * SMDIM, TB), TB, 0, stream>>>(Wc, Wc_p, HID, SMDIM);

    // ---- GIN layer 0 (K=128 input) ----
    zerof_kernel<<<cdiv(N_NODES * IN_CH, TB), TB, 0, stream>>>(agg, N_NODES * IN_CH);
    gather0_kernel<<<cdiv(N_EDGES * 32, TB), TB, 0, stream>>>(x, src, dst, agg);
    combine0_kernel<<<cdiv(N_NODES * IN_CH, TB), TB, 0, stream>>>(x, agg, a16);
    {
        int waves = (N_NODES / 32) * (HID / 64);
        wmma_gemm_kernel<<<cdiv(waves * 32, TB), TB, 0, stream>>>(
            a16, w0a_p, b0a, b16, nullptr, N_NODES, IN_CH, HID, /*relu*/1);
        wmma_gemm_kernel<<<cdiv(waves * 32, TB), TB, 0, stream>>>(
            b16, w0b_p, b0b, h16, nullptr, N_NODES, HID, HID, /*none*/0);
    }

    // ---- GIN layers 1..2 (relu -> conv) ----
    for (int i = 0; i < 2; ++i) {
        const h16_t* wa = (i == 0) ? wra0_p : wra1_p;
        const h16_t* wb = (i == 0) ? wrb0_p : wrb1_p;
        const float* ba = bra + i * HID;
        const float* bb = brb + i * HID;
        zerof_kernel<<<cdiv(N_NODES * HID, TB), TB, 0, stream>>>(agg, N_NODES * HID);
        gather_h_kernel<<<cdiv(N_EDGES * 32, TB), TB, 0, stream>>>(h16, src, dst, agg);
        combine_h_kernel<<<cdiv(N_NODES * HID, TB), TB, 0, stream>>>(h16, agg, a16);
        int waves = (N_NODES / 32) * (HID / 64);
        wmma_gemm_kernel<<<cdiv(waves * 32, TB), TB, 0, stream>>>(
            a16, wa, ba, b16, nullptr, N_NODES, HID, HID, 1);
        wmma_gemm_kernel<<<cdiv(waves * 32, TB), TB, 0, stream>>>(
            b16, wb, bb, h16, nullptr, N_NODES, HID, HID, 0);
    }

    // ---- pooling + set-rep classifier ----
    pool_kernel<<<cdiv(NCOMP * HID, TB), TB, 0, stream>>>(h16, Xp);
    {
        int waves = (NCOMP / 32) * (SMDIM / 64);
        wmma_gemm_kernel<<<cdiv(waves * 32, TB), TB, 0, stream>>>(
            Xp, Wc_p, nullptr, nullptr, tcls, NCOMP, HID, SMDIM, /*leaky*/2);
    }
    setred_kernel<<<cdiv(BATCH * SDIM, TB), TB, 0, stream>>>(tcls, tsum);
    fc1_kernel<<<cdiv(BATCH * 32, TB), TB, 0, stream>>>(tsum, fc1_w, fc1_b, t1);
    bnstats_kernel<<<32, 256, 0, stream>>>(t1, stats);
    head_kernel<<<cdiv(BATCH, TB), TB, 0, stream>>>(t1, stats, bn_g, bn_b, fc2_w, fc2_b, out);

    (void)in_sizes; (void)n_in; (void)out_size; (void)ws_size;
}
